// MultiHeadAttention_36962488549641
// MI455X (gfx1250) — matrix operand order repaired and checked by reading, not compiled, not run on hardware
//
#include <hip/hip_runtime.h>
#include <hip/hip_bf16.h>

// ---------------------------------------------------------------------------
// MHA forward for MI455X (gfx1250, wave32, WMMA).
// All matmuls (QKV proj, QK^T, P*V, softmax row-sum, out proj) use
// v_wmma_f32_16x16x32_bf16.
// ---------------------------------------------------------------------------

typedef __attribute__((ext_vector_type(16))) __bf16 v16bf;
typedef __attribute__((ext_vector_type(8)))  float  v8f;
typedef __attribute__((ext_vector_type(4)))  float  f32x4;

#define D_MODEL 1024
#define SEQ     2048
#define NHEAD   16
#define DK      64
#define MTOT    4096   // B*S

// Load a 32-byte WMMA fragment as two 16-byte chunks.
__device__ __forceinline__ v16bf load_frag(const __bf16* p0, const __bf16* p1) {
    v16bf f;
    f32x4* d = reinterpret_cast<f32x4*>(&f);
    d[0] = *reinterpret_cast<const f32x4*>(p0);
    d[1] = *reinterpret_cast<const f32x4*>(p1);
    return f;
}

__device__ __forceinline__ v8f wmma_bf16(v16bf a, v16bf b, v8f c) {
    return __builtin_amdgcn_wmma_f32_16x16x32_bf16(
        false, a, false, b, (short)0, c, false, false);
}

// ---------------------------------------------------------------------------
// f32 -> bf16 conversion, 8 elements per thread
// ---------------------------------------------------------------------------
__global__ __launch_bounds__(256)
void cvt_f32_bf16(const float* __restrict__ src, __bf16* __restrict__ dst, int n) {
    const int i8 = (blockIdx.x * 256 + threadIdx.x) * 8;
    if (i8 + 8 <= n) {
        f32x4 a = *reinterpret_cast<const f32x4*>(src + i8);
        f32x4 b = *reinterpret_cast<const f32x4*>(src + i8 + 4);
        union { f32x4 v; __bf16 h[8]; } u;
        #pragma unroll
        for (int e = 0; e < 4; ++e) { u.h[e] = (__bf16)a[e]; u.h[4 + e] = (__bf16)b[e]; }
        *reinterpret_cast<f32x4*>(dst + i8) = u.v;
    }
}

// ---------------------------------------------------------------------------
// C[M,N] = (A[M,K] * B[N,K]^T + bias[N]) * oscale
// A,B bf16 row-major (K contiguous). Output bf16 or f32.
// Block: 256 threads = 8 waves. Block tile 128(M) x 128(N), K chunk 32.
// Waves 4(M) x 2(N); wave tile 32x64 = 2x4 WMMA accumulators.
// ---------------------------------------------------------------------------
template<bool F32OUT>
__global__ __launch_bounds__(256)
void gemm_bt(const __bf16* __restrict__ A, const __bf16* __restrict__ Bm,
             const float* __restrict__ bias,
             __bf16* __restrict__ outb, float* __restrict__ outf,
             int M, int N, int K, float oscale) {
    __shared__ __align__(16) __bf16 sA[128 * 32];
    __shared__ __align__(16) __bf16 sB[128 * 32];

    const int tid  = threadIdx.x;
    const int wid  = tid >> 5;
    const int lane = tid & 31;
    const int l16  = lane & 15;
    const int hi   = (lane >> 4) & 1;
    const int klo  = hi ? 8 : 0;     // A-fragment k low offset (interleaved layout)
    const int khal = hi ? 16 : 0;    // unused (B fragments also use klo, interleaved)

    const int m0 = blockIdx.y * 128;
    const int n0 = blockIdx.x * 128;
    const int m_off = (wid >> 1) * 32;   // 4 waves along M, 32 rows each
    const int n_off = (wid & 1) * 64;    // 2 waves along N, 64 cols each

    // staging coords: 4 threads per 32-element row, 64 rows per round
    const int sr = tid >> 2;
    const int sc = (tid & 3) * 8;

    v8f acc[2][4] = {};

    for (int kc = 0; kc < K; kc += 32) {
        // ---- stage A (128x32) and B (128x32) into LDS ----
        *reinterpret_cast<f32x4*>(&sA[sr * 32 + sc]) =
            *reinterpret_cast<const f32x4*>(A + (size_t)(m0 + sr) * K + kc + sc);
        *reinterpret_cast<f32x4*>(&sA[(sr + 64) * 32 + sc]) =
            *reinterpret_cast<const f32x4*>(A + (size_t)(m0 + sr + 64) * K + kc + sc);
        *reinterpret_cast<f32x4*>(&sB[sr * 32 + sc]) =
            *reinterpret_cast<const f32x4*>(Bm + (size_t)(n0 + sr) * K + kc + sc);
        *reinterpret_cast<f32x4*>(&sB[(sr + 64) * 32 + sc]) =
            *reinterpret_cast<const f32x4*>(Bm + (size_t)(n0 + sr + 64) * K + kc + sc);
        if (kc + 32 < K) {  // prefetch next K chunk
            __builtin_prefetch(A + (size_t)(m0 + sr) * K + kc + 32 + sc, 0, 1);
            __builtin_prefetch(Bm + (size_t)(n0 + sr) * K + kc + 32 + sc, 0, 1);
        }
        __syncthreads();

        // ---- fragments (all loaded up-front so WMMAs issue back-to-back) ----
        v16bf af[2], bf[4];
        #pragma unroll
        for (int mt = 0; mt < 2; ++mt) {
            const __bf16* p = &sA[(m_off + mt * 16 + l16) * 32];
            af[mt] = load_frag(p + klo, p + klo + 16);
        }
        #pragma unroll
        for (int nt = 0; nt < 4; ++nt) {
            const __bf16* p = &sB[(n_off + nt * 16 + l16) * 32];
            bf[nt] = load_frag(p + klo, p + klo + 16);
        }
        #pragma unroll
        for (int mt = 0; mt < 2; ++mt)
            #pragma unroll
            for (int nt = 0; nt < 4; ++nt)
                acc[mt][nt] = wmma_bf16(af[mt], bf[nt], acc[mt][nt]);
        __syncthreads();
    }

    // ---- epilogue: bias + scale + store ----
    #pragma unroll
    for (int nt = 0; nt < 4; ++nt) {
        const int col = n0 + n_off + nt * 16 + l16;
        const float bv = bias[col];
        #pragma unroll
        for (int mt = 0; mt < 2; ++mt) {
            #pragma unroll
            for (int i = 0; i < 8; ++i) {
                const int row = m0 + m_off + mt * 16 + i + hi * 8;
                const float cv = (acc[mt][nt][i] + bv) * oscale;
                if constexpr (F32OUT)
                    outf[(size_t)row * N + col] = cv;
                else
                    outb[(size_t)row * N + col] = (__bf16)cv;
            }
        }
    }
}

// ---------------------------------------------------------------------------
// Flash attention. Q,K,V stored as bf16 [4096, 1024] (head h at cols h*64..).
// Q is PRE-SCALED (in its projection GEMM) by (1/sqrt(64))*log2(e): exp2 softmax.
// Row sums of P are computed with a WMMA against a ones matrix (no shuffle tree).
// Grid: (S/64, B*H). Block: 128 threads = 4 waves; each wave owns 16 queries.
// V is staged TRANSPOSED + key-pair-packed so P*V B-frags are ds_load_b128s.
// ---------------------------------------------------------------------------
__global__ __launch_bounds__(128)
void flash_attn(const __bf16* __restrict__ Q, const __bf16* __restrict__ Km,
                const __bf16* __restrict__ V, __bf16* __restrict__ O) {
    __shared__ __align__(16) __bf16 sK[32 * 64];    // [key][d]
    __shared__ __align__(16) __bf16 sVt[64 * 32];   // [d][key]  (transposed)
    __shared__ __align__(16) __bf16 sP[4][16 * 32];

    const int tid  = threadIdx.x;
    const int wid  = tid >> 5;
    const int lane = tid & 31;
    const int l16  = lane & 15;
    const int hi   = (lane >> 4) & 1;
    const int klo  = hi ? 8 : 0;
    const int khal = hi ? 16 : 0;

    const int bh = blockIdx.y;
    const int b  = bh >> 4;
    const int h  = bh & 15;
    const int q0 = blockIdx.x * 64 + wid * 16;     // query index within batch

    const size_t bhoff = (size_t)b * SEQ * D_MODEL + (size_t)h * DK;
    const __bf16* Qg = Q + bhoff;
    const __bf16* Kg = Km + bhoff;
    const __bf16* Vg = V + bhoff;

    // ---- load Q fragments once (A layout); already scaled by proj GEMM ----
    v16bf aQ[2];
    {
        const __bf16* qrow = Qg + (size_t)(q0 + l16) * D_MODEL;
        #pragma unroll
        for (int j = 0; j < 2; ++j)
            aQ[j] = load_frag(qrow + j * 32 + klo, qrow + j * 32 + klo + 16);
    }

    // ones B-matrix for row-sum WMMA
    v16bf ones;
    #pragma unroll
    for (int e = 0; e < 16; ++e) ones[e] = (__bf16)1.0f;

    float mrow[8], lrow[8];
    #pragma unroll
    for (int i = 0; i < 8; ++i) { mrow[i] = -1e30f; lrow[i] = 0.0f; }
    v8f o[4] = {};

    // K staging coords: 32 rows x 64 cols, 16 elems per thread
    const int vr = tid >> 2;
    const int vc = (tid & 3) * 16;
    // V transpose-staging coords: key pair kp, 8 d values per thread
    const int kp    = tid >> 3;          // 0..15 -> keys 2kp, 2kp+1
    const int dbase = (tid & 7) * 8;     // 0..56

    for (int kt = 0; kt < SEQ / 32; ++kt) {
        const int k0 = kt * 32;

        // ---- stage K tile (32 x 64, row-major) ----
        *reinterpret_cast<f32x4*>(&sK[vr * 64 + vc]) =
            *reinterpret_cast<const f32x4*>(Kg + (size_t)(k0 + vr) * D_MODEL + vc);
        *reinterpret_cast<f32x4*>(&sK[vr * 64 + vc + 8]) =
            *reinterpret_cast<const f32x4*>(Kg + (size_t)(k0 + vr) * D_MODEL + vc + 8);

        // ---- stage V tile transposed: sVt[d][key], key pairs packed ----
        {
            union { f32x4 v; unsigned short s[8]; } r0, r1;
            r0.v = *reinterpret_cast<const f32x4*>(
                Vg + (size_t)(k0 + 2 * kp) * D_MODEL + dbase);
            r1.v = *reinterpret_cast<const f32x4*>(
                Vg + (size_t)(k0 + 2 * kp + 1) * D_MODEL + dbase);
            unsigned int* sVtw = reinterpret_cast<unsigned int*>(sVt);
            #pragma unroll
            for (int e = 0; e < 8; ++e) {
                const unsigned int packed =
                    (unsigned int)r0.s[e] | ((unsigned int)r1.s[e] << 16);
                sVtw[(dbase + e) * 16 + kp] = packed;   // = &sVt[(d)*32 + 2kp]
            }
        }
        if (kt + 1 < SEQ / 32) {  // prefetch next tile
            __builtin_prefetch(Kg + (size_t)(k0 + 32 + vr) * D_MODEL + vc, 0, 1);
            __builtin_prefetch(Vg + (size_t)(k0 + 32 + 2 * kp) * D_MODEL + dbase, 0, 1);
        }
        __syncthreads();

        // ---- scores: 16 queries x 32 keys (already in exp2 domain) ----
        v16bf bk[2][2];
        #pragma unroll
        for (int nt = 0; nt < 2; ++nt)
            #pragma unroll
            for (int j = 0; j < 2; ++j) {
                const __bf16* p = &sK[(nt * 16 + l16) * 64 + j * 32];
                bk[nt][j] = load_frag(p + klo, p + klo + 16);
            }
        v8f scv[2] = {};
        #pragma unroll
        for (int nt = 0; nt < 2; ++nt)
            #pragma unroll
            for (int j = 0; j < 2; ++j)
                scv[nt] = wmma_bf16(aQ[j], bk[nt][j], scv[nt]);

        // ---- online softmax max + P (row stats across 16 lanes per half) ----
        float alpha[8];
        #pragma unroll
        for (int i = 0; i < 8; ++i) {
            float mx = fmaxf(scv[0][i], scv[1][i]);
            #pragma unroll
            for (int off = 1; off < 16; off <<= 1)
                mx = fmaxf(mx, __shfl_xor(mx, off, 16));
            const float mnew = fmaxf(mrow[i], mx);
            alpha[i] = exp2f(mrow[i] - mnew);
            mrow[i] = mnew;

            const float p0 = exp2f(scv[0][i] - mnew);
            const float p1 = exp2f(scv[1][i] - mnew);
            const int row = i + hi * 8;
            sP[wid][row * 32 +      l16] = (__bf16)p0;
            sP[wid][row * 32 + 16 + l16] = (__bf16)p1;
        }

        // ---- rescale running output ----
        #pragma unroll
        for (int dt = 0; dt < 4; ++dt)
            #pragma unroll
            for (int i = 0; i < 8; ++i)
                o[dt][i] *= alpha[i];

        // ---- P (A layout, from per-wave LDS; DS ops are in-order per wave) ----
        const __bf16* pp = &sP[wid][l16 * 32];
        v16bf pa = load_frag(pp + klo, pp + klo + 16);

        // ---- row sums of P via WMMA against ones ----
        v8f srow = wmma_bf16(pa, ones, v8f{});
        #pragma unroll
        for (int i = 0; i < 8; ++i)
            lrow[i] = lrow[i] * alpha[i] + srow[i];

        // ---- P * V: B fragments are contiguous rows of sVt ----
        v16bf bv[4];
        #pragma unroll
        for (int dt = 0; dt < 4; ++dt) {
            const __bf16* p = &sVt[(dt * 16 + l16) * 32];
            bv[dt] = load_frag(p + klo, p + klo + 16);
        }
        #pragma unroll
        for (int dt = 0; dt < 4; ++dt)
            o[dt] = wmma_bf16(pa, bv[dt], o[dt]);
        __syncthreads();
    }

    // ---- normalize and store ----
    float invl[8];
    #pragma unroll
    for (int i = 0; i < 8; ++i) invl[i] = 1.0f / lrow[i];

    __bf16* Ob = O + bhoff + (size_t)q0 * D_MODEL;
    #pragma unroll
    for (int dt = 0; dt < 4; ++dt)
        #pragma unroll
        for (int i = 0; i < 8; ++i)
            Ob[(size_t)(i + hi * 8) * D_MODEL + dt * 16 + l16] =
                (__bf16)(o[dt][i] * invl[i]);
}

// ---------------------------------------------------------------------------
// Launch
// ---------------------------------------------------------------------------
extern "C" void kernel_launch(void* const* d_in, const int* in_sizes, int n_in,
                              void* d_out, int out_size, void* d_ws, size_t ws_size,
                              hipStream_t stream) {
    const float* x  = (const float*)d_in[0];
    const float* Wq = (const float*)d_in[1];
    const float* bq = (const float*)d_in[2];
    const float* Wk = (const float*)d_in[3];
    const float* bk = (const float*)d_in[4];
    const float* Wv = (const float*)d_in[5];
    const float* bv = (const float*)d_in[6];
    const float* Wo = (const float*)d_in[7];
    const float* bo = (const float*)d_in[8];
    float* out = (float*)d_out;

    const size_t MB = 1u << 20;
    if (ws_size < 48 * MB) return;  // need 48MB scratch

    char* ws = (char*)d_ws;
    __bf16* xb  = (__bf16*)(ws);             //  8MB: x bf16 [4096,1024]
    __bf16* wqb = (__bf16*)(ws +  8 * MB);   //  2MB each
    __bf16* wkb = (__bf16*)(ws + 10 * MB);
    __bf16* wvb = (__bf16*)(ws + 12 * MB);
    __bf16* wob = (__bf16*)(ws + 14 * MB);
    __bf16* Qb  = (__bf16*)(ws + 16 * MB);   //  8MB each
    __bf16* Kb  = (__bf16*)(ws + 24 * MB);
    __bf16* Vb  = (__bf16*)(ws + 32 * MB);
    __bf16* Ab  = (__bf16*)(ws + 40 * MB);

    // 1) convert inputs to bf16 (8 elems/thread)
    const int nx = MTOT * D_MODEL;       // 4,194,304
    const int nw = D_MODEL * D_MODEL;    // 1,048,576
    cvt_f32_bf16<<<nx / (256 * 8), 256, 0, stream>>>(x,  xb,  nx);
    cvt_f32_bf16<<<nw / (256 * 8), 256, 0, stream>>>(Wq, wqb, nw);
    cvt_f32_bf16<<<nw / (256 * 8), 256, 0, stream>>>(Wk, wkb, nw);
    cvt_f32_bf16<<<nw / (256 * 8), 256, 0, stream>>>(Wv, wvb, nw);
    cvt_f32_bf16<<<nw / (256 * 8), 256, 0, stream>>>(Wo, wob, nw);

    // 2) Q/K/V projections (bf16 out); Q pre-scaled into exp2 softmax domain
    const float QK_SCALE = 0.125f * 1.44269504088896f;  // 1/sqrt(64) * log2(e)
    dim3 ggrid(D_MODEL / 128, MTOT / 128);  // (8, 32)
    gemm_bt<false><<<ggrid, 256, 0, stream>>>(xb, wqb, bq, Qb, nullptr,
                                              MTOT, D_MODEL, D_MODEL, QK_SCALE);
    gemm_bt<false><<<ggrid, 256, 0, stream>>>(xb, wkb, bk, Kb, nullptr,
                                              MTOT, D_MODEL, D_MODEL, 1.0f);
    gemm_bt<false><<<ggrid, 256, 0, stream>>>(xb, wvb, bv, Vb, nullptr,
                                              MTOT, D_MODEL, D_MODEL, 1.0f);

    // 3) flash attention -> Ab (bf16 [4096,1024], head-major cols)
    dim3 agrid(SEQ / 64, 2 * NHEAD);  // (32, 32)
    flash_attn<<<agrid, 128, 0, stream>>>(Qb, Kb, Vb, Ab);

    // 4) output projection (f32 out)
    gemm_bt<true><<<ggrid, 256, 0, stream>>>(Ab, wob, bo, nullptr, out,
                                             MTOT, D_MODEL, D_MODEL, 1.0f);
}
